// GraphTransformerLayer_69398081569511
// MI455X (gfx1250) — matrix operand order repaired and checked by reading, not compiled, not run on hardware
//
#include <hip/hip_runtime.h>
#include <hip/hip_bf16.h>
#include <math.h>

// ---------------------------------------------------------------------------
// gfx1250 WMMA (wave32): D(16x16 f32) = A(16x32 bf16) x B(32x16 bf16) + C
// ---------------------------------------------------------------------------
typedef __attribute__((ext_vector_type(16))) __bf16        v16bf;
typedef __attribute__((ext_vector_type(8)))  unsigned int  v8u;
typedef __attribute__((ext_vector_type(8)))  float         v8f;

__device__ __forceinline__ unsigned int f2bf(float f) {
    // round-to-nearest-even f32 -> bf16 bits
    unsigned int u = __builtin_bit_cast(unsigned int, f);
    unsigned int r = u + 0x7FFFu + ((u >> 16) & 1u);
    return r >> 16;
}

__device__ __forceinline__ v8f wmma_bf16(v16bf a, v16bf b, v8f c) {
    // (neg_a, A, neg_b, B, c_mod, C, reuse_a, reuse_b)
    return __builtin_amdgcn_wmma_f32_16x16x32_bf16(false, a, false, b,
                                                   (short)0, c, false, false);
}

__device__ __forceinline__ v16bf frag_from_2x128(uint4 lo, uint4 hi) {
    v8u u;
    u[0] = lo.x; u[1] = lo.y; u[2] = lo.z; u[3] = lo.w;
    u[4] = hi.x; u[5] = hi.y; u[6] = hi.z; u[7] = hi.w;
    return __builtin_bit_cast(v16bf, u);
}

// A fragment: 8 consecutive dwords from the fragment-ordered LDS tile.
__device__ __forceinline__ v16bf load_a_lds(const unsigned int* xt,
                                            int mrow, int half, int kk) {
    const unsigned int* p = xt + ((mrow * 2 + half) * 32 + kk * 8);
    uint4 lo = ((const uint4*)p)[0];
    uint4 hi = ((const uint4*)p)[1];
    return frag_from_2x128(lo, hi);
}

// B fragment: 8 consecutive dwords from the pre-packed weight buffer.
__device__ __forceinline__ v16bf load_b_packed(const unsigned int* __restrict__ Wp,
                                               int ct, int kk, int lane) {
    const unsigned int* p = Wp + ((((ct * 4 + kk) * 32) + lane) << 3);
    uint4 lo = ((const uint4*)p)[0];
    uint4 hi = ((const uint4*)p)[1];
    return frag_from_2x128(lo, hi);
}

// Fill a fragment-ordered 16x128 bf16 tile in LDS from row-major bf16 global
// memory using gfx1250 async-to-LDS 16B copies (ASYNCcnt-tracked).
// Thread t (0..255) copies chunk (row r = t>>4, k0 = (t&15)*8).
__device__ __forceinline__ void async_tile_load(unsigned int* xt /*LDS*/,
        const unsigned short* __restrict__ srcbf, int row0, int n, int tid) {
    int r   = tid >> 4;
    int k0  = (tid & 15) * 8;               // 8 bf16 = 16 bytes, contiguous
    int row = row0 + r; if (row >= n) row = n - 1;   // clamp (stores masked)
    int kk   = k0 >> 5;
    int rem  = k0 & 31;                     // 0,8,16,24
    int half = (rem >> 3) & 1;              // 0,1,0,1
    int sub  = (rem & 16) ? 4 : 0;          // VGPR 0-3 vs 4-7 group
    unsigned lds_addr = (unsigned)(uintptr_t)(xt + ((r * 2 + half) * 32 + kk * 8 + sub));
    unsigned long long gaddr =
        (unsigned long long)(uintptr_t)(srcbf + (size_t)row * 128 + k0);
    asm volatile("global_load_async_to_lds_b128 %0, %1, off"
                 :: "v"(lds_addr), "v"(gaddr) : "memory");
}

__device__ __forceinline__ void wait_async_and_barrier() {
    asm volatile("s_wait_asynccnt 0x0" ::: "memory");
    __syncthreads();
}

// ---------------------------------------------------------------------------
// Pre-pack kernels (run once per launch; tiny)
// ---------------------------------------------------------------------------

// x (f32 row-major) -> bf16 row-major, 4 elements / thread
__global__ __launch_bounds__(256) void cvt_bf16_kernel(
    const float* __restrict__ src, unsigned int* __restrict__ dst, long long n4)
{
    long long t = (long long)blockIdx.x * 256 + threadIdx.x;
    if (t >= n4) return;
    float4 f = ((const float4*)src)[t];
    uint2 o;
    o.x = f2bf(f.x) | (f2bf(f.y) << 16);
    o.y = f2bf(f.z) | (f2bf(f.w) << 16);
    ((uint2*)dst)[t] = o;
}

// W (f32 row-major 128x128) -> fragment-ready packed bf16:
//   Wp[((ct*4+kk)*32+lane)*8 + v] = {bf16 W[k,colg], bf16 W[k+1,colg]},
//   colg = ct*16 + (lane&15), k = kk*32 + (v>>2)*16 + (lane>>4)*8 + 2(v&3).
__global__ __launch_bounds__(256) void pack_w_kernel(
    const float* __restrict__ W, unsigned int* __restrict__ Wp)
{
    int t = blockIdx.x * 256 + threadIdx.x;     // 0..8191
    if (t >= 8192) return;
    int v    = t & 7;
    int lane = (t >> 3) & 31;
    int kk   = (t >> 8) & 3;
    int ct   = t >> 10;
    int colg = ct * 16 + (lane & 15);
    int k    = kk * 32 + (v >> 2) * 16 + (lane >> 4) * 8 + 2 * (v & 3);
    Wp[t] = f2bf(W[k * 128 + colg]) | (f2bf(W[(k + 1) * 128 + colg]) << 16);
}

// ---------------------------------------------------------------------------
// Kernel 1: fused QKV projection.  Block = 256 thr (8 waves) = one 16-row tile;
// wave w owns column tile w of Q, K, V: 12 WMMAs/wave, all operands b128 loads.
// ---------------------------------------------------------------------------
__global__ __launch_bounds__(256) void qkv_kernel(
    const unsigned short* __restrict__ xbf,
    const unsigned int* __restrict__ Wqp, const unsigned int* __restrict__ Wkp,
    const unsigned int* __restrict__ Wvp,
    const float* __restrict__ bq, const float* __restrict__ bk,
    const float* __restrict__ bv,
    float* __restrict__ Q, float* __restrict__ K, float* __restrict__ V, int n)
{
    __shared__ __align__(16) unsigned int xt[16 * 64];   // fragment-ordered tile
    const int row0 = blockIdx.x * 16;
    const int tid  = threadIdx.x;

    async_tile_load(xt, xbf, row0, n, tid);
    wait_async_and_barrier();

    const int wave = tid >> 5;          // 0..7 -> column tile
    const int lane = tid & 31;
    const int ct   = wave;
    const int half = lane >> 4;
    const int mrow = lane & 15;
    const int colg = ct * 16 + mrow;

    v8f accq = {}, acck = {}, accv = {};
#pragma unroll
    for (int kk = 0; kk < 4; ++kk) {
        v16bf a  = load_a_lds(xt, mrow, half, kk);
        v16bf fq = load_b_packed(Wqp, ct, kk, lane);
        v16bf fk = load_b_packed(Wkp, ct, kk, lane);
        v16bf fv = load_b_packed(Wvp, ct, kk, lane);
        accq = wmma_bf16(a, fq, accq);
        acck = wmma_bf16(a, fk, acck);
        accv = wmma_bf16(a, fv, accv);
    }

    const float biq = bq[colg], bik = bk[colg], biv = bv[colg];
#pragma unroll
    for (int i = 0; i < 8; ++i) {
        int row = row0 + i + half * 8;            // C/D: VGPR i -> M = i + half*8
        if (row < n) {
            size_t idx = (size_t)row * 128 + colg;
            Q[idx] = accq[i] + biq;
            K[idx] = acck[i] + bik;
            V[idx] = accv[i] + biv;
        }
    }
}

// ---------------------------------------------------------------------------
// Kernel 2: per-(edge,head) score + scatter.  One thread per (e,h).
// Entire Q/K/V/wV working set (~130 MB) is L2-resident on MI455X (192 MB L2).
// ---------------------------------------------------------------------------
__global__ __launch_bounds__(256) void edge_kernel(
    const int* __restrict__ src, const int* __restrict__ dst,
    const float* __restrict__ Q, const float* __restrict__ K,
    const float* __restrict__ V,
    float* __restrict__ wV, float* __restrict__ z, int E)
{
    int t = blockIdx.x * blockDim.x + threadIdx.x;
    if (t >= E * 8) return;
    const int e = t >> 3, h = t & 7;
    const int s = src[e], d = dst[e];

    const float4* kp = (const float4*)(K + (size_t)s * 128 + h * 16);
    const float4* qp = (const float4*)(Q + (size_t)d * 128 + h * 16);
    const float4* vp = (const float4*)(V + (size_t)s * 128 + h * 16);
    __builtin_prefetch(vp, 0, 1);       // global_prefetch_b8: warm V while scoring

    float acc = 0.0f;
#pragma unroll
    for (int i = 0; i < 4; ++i) {
        float4 a = kp[i], b = qp[i];
        acc += a.x * b.x + a.y * b.y + a.z * b.z + a.w * b.w;
    }
    acc *= 0.25f;                                    // 1/sqrt(16)
    acc = fminf(5.0f, fmaxf(-5.0f, acc));
    const float sc = __expf(acc);

    atomicAdd(z + (size_t)d * 8 + h, sc);
    float* wp = wV + (size_t)d * 128 + h * 16;
#pragma unroll
    for (int i = 0; i < 4; ++i) {
        float4 v = vp[i];
        atomicAdd(wp + 4 * i + 0, sc * v.x);
        atomicAdd(wp + 4 * i + 1, sc * v.y);
        atomicAdd(wp + 4 * i + 2, sc * v.z);
        atomicAdd(wp + 4 * i + 3, sc * v.w);
    }
}

// ---------------------------------------------------------------------------
// Kernel 3: attn normalize + residual + LayerNorm1.  One wave32 per node.
// ---------------------------------------------------------------------------
__global__ __launch_bounds__(256) void attn_ln1_kernel(
    const float* __restrict__ x, const float* __restrict__ wV,
    const float* __restrict__ z,
    const float* __restrict__ gamma, const float* __restrict__ beta,
    float* __restrict__ h1, int n)
{
    int node = (blockIdx.x * blockDim.x + threadIdx.x) >> 5;
    int lane = threadIdx.x & 31;
    if (node >= n) return;

    const int head = (lane * 4) >> 4;
    const float zz = z[(size_t)node * 8 + head] + 1e-3f;

    float4 xv = ((const float4*)(x  + (size_t)node * 128))[lane];
    float4 wv = ((const float4*)(wV + (size_t)node * 128))[lane];
    float v0 = xv.x + wv.x / zz, v1 = xv.y + wv.y / zz;
    float v2 = xv.z + wv.z / zz, v3 = xv.w + wv.w / zz;

    float sum = v0 + v1 + v2 + v3;
#pragma unroll
    for (int off = 16; off; off >>= 1) sum += __shfl_xor(sum, off, 32);
    const float mu = sum * (1.0f / 128.0f);

    float d0 = v0 - mu, d1 = v1 - mu, d2 = v2 - mu, d3 = v3 - mu;
    float var = d0 * d0 + d1 * d1 + d2 * d2 + d3 * d3;
#pragma unroll
    for (int off = 16; off; off >>= 1) var += __shfl_xor(var, off, 32);
    const float r = rsqrtf(var * (1.0f / 128.0f) + 1e-5f);

    float4 gv = ((const float4*)gamma)[lane];
    float4 bv = ((const float4*)beta)[lane];
    float4 o;
    o.x = d0 * r * gv.x + bv.x;  o.y = d1 * r * gv.y + bv.y;
    o.z = d2 * r * gv.z + bv.z;  o.w = d3 * r * gv.w + bv.w;
    ((float4*)(h1 + (size_t)node * 128))[lane] = o;
}

// ---------------------------------------------------------------------------
// Kernel 4: LayerNorm2, emits bf16 directly (gbf) for the FFN WMMA.
// ---------------------------------------------------------------------------
__global__ __launch_bounds__(256) void ln2_kernel(
    const float* __restrict__ h1,
    const float* __restrict__ gamma, const float* __restrict__ beta,
    unsigned short* __restrict__ gbf, int n)
{
    int node = (blockIdx.x * blockDim.x + threadIdx.x) >> 5;
    int lane = threadIdx.x & 31;
    if (node >= n) return;

    float4 hv = ((const float4*)(h1 + (size_t)node * 128))[lane];
    float sum = hv.x + hv.y + hv.z + hv.w;
#pragma unroll
    for (int off = 16; off; off >>= 1) sum += __shfl_xor(sum, off, 32);
    const float mu = sum * (1.0f / 128.0f);

    float d0 = hv.x - mu, d1 = hv.y - mu, d2 = hv.z - mu, d3 = hv.w - mu;
    float var = d0 * d0 + d1 * d1 + d2 * d2 + d3 * d3;
#pragma unroll
    for (int off = 16; off; off >>= 1) var += __shfl_xor(var, off, 32);
    const float r = rsqrtf(var * (1.0f / 128.0f) + 1e-5f);

    float4 gv = ((const float4*)gamma)[lane];
    float4 bv = ((const float4*)beta)[lane];
    uint2 o;
    o.x = f2bf(d0 * r * gv.x + bv.x) | (f2bf(d1 * r * gv.y + bv.y) << 16);
    o.y = f2bf(d2 * r * gv.z + bv.z) | (f2bf(d3 * r * gv.w + bv.w) << 16);
    ((uint2*)(gbf + (size_t)node * 128))[lane] = o;
}

// ---------------------------------------------------------------------------
// Kernel 5: FFN  out = h1 + relu(g @ Wo + bo).  Same WMMA tiling as QKV.
// ---------------------------------------------------------------------------
__global__ __launch_bounds__(256) void ffn_kernel(
    const unsigned short* __restrict__ gbf, const float* __restrict__ h1,
    const unsigned int* __restrict__ Wop, const float* __restrict__ bo,
    float* __restrict__ out, int n)
{
    __shared__ __align__(16) unsigned int gt[16 * 64];
    const int row0 = blockIdx.x * 16;
    const int tid  = threadIdx.x;

    async_tile_load(gt, gbf, row0, n, tid);
    wait_async_and_barrier();

    const int wave = tid >> 5;
    const int lane = tid & 31;
    const int ct   = wave;
    const int half = lane >> 4;
    const int mrow = lane & 15;
    const int colg = ct * 16 + mrow;

    v8f acc = {};
#pragma unroll
    for (int kk = 0; kk < 4; ++kk) {
        v16bf a = load_a_lds(gt, mrow, half, kk);
        v16bf b = load_b_packed(Wop, ct, kk, lane);
        acc = wmma_bf16(a, b, acc);
    }

    const float bias = bo[colg];
#pragma unroll
    for (int i = 0; i < 8; ++i) {
        int row = row0 + i + half * 8;
        if (row < n) {
            size_t idx = (size_t)row * 128 + colg;
            float t = acc[i] + bias;
            out[idx] = h1[idx] + fmaxf(t, 0.0f);
        }
    }
}

// ---------------------------------------------------------------------------
// Host-side launcher (graph-capture safe: only async ops on `stream`).
// ---------------------------------------------------------------------------
extern "C" void kernel_launch(void* const* d_in, const int* in_sizes, int n_in,
                              void* d_out, int out_size, void* d_ws, size_t ws_size,
                              hipStream_t stream)
{
    (void)n_in; (void)out_size; (void)ws_size;
    const float* x      = (const float*)d_in[0];
    const int*   src    = (const int*)  d_in[1];
    const int*   dst    = (const int*)  d_in[2];
    const float* Wq     = (const float*)d_in[3];
    const float* bq     = (const float*)d_in[4];
    const float* Wk     = (const float*)d_in[5];
    const float* bk     = (const float*)d_in[6];
    const float* Wv     = (const float*)d_in[7];
    const float* bv     = (const float*)d_in[8];
    const float* Wo     = (const float*)d_in[9];
    const float* bo     = (const float*)d_in[10];
    const float* gamma1 = (const float*)d_in[11];
    const float* beta1  = (const float*)d_in[12];
    const float* gamma2 = (const float*)d_in[13];
    const float* beta2  = (const float*)d_in[14];

    const int n = in_sizes[0] / 128;
    const int E = in_sizes[1];
    const size_t ND = (size_t)n * 128;

    // Workspace layout: f32 Q|K|V|wV|h1|z, then bf16 xbf|gbf, then packed W.
    float* Q  = (float*)d_ws;
    float* K  = Q  + ND;
    float* V  = K  + ND;
    float* wV = V  + ND;
    float* h1 = wV + ND;
    float* z  = h1 + ND;                       // n*8 floats
    unsigned short* xbf = (unsigned short*)(z + (size_t)n * 8);
    unsigned short* gbf = xbf + ND;
    unsigned int* Wqp = (unsigned int*)(gbf + ND);
    unsigned int* Wkp = Wqp + 8192;
    unsigned int* Wvp = Wkp + 8192;
    unsigned int* Wop = Wvp + 8192;

    // Accumulators must be re-zeroed every call (graph replays don't re-poison).
    hipMemsetAsync(wV, 0, ND * sizeof(float), stream);
    hipMemsetAsync(z,  0, (size_t)n * 8 * sizeof(float), stream);

    const long long n4   = (long long)(ND / 4);
    const int cvtBlocks  = (int)((n4 + 255) / 256);
    const int rowTiles   = (n + 15) / 16;
    const int edgeBlocks = (E * 8 + 255) / 256;
    const int nodeBlocks = (n * 32 + 255) / 256;   // one wave32 per node

    cvt_bf16_kernel<<<cvtBlocks, 256, 0, stream>>>(x, (unsigned int*)xbf, n4);
    pack_w_kernel<<<32, 256, 0, stream>>>(Wq, Wqp);
    pack_w_kernel<<<32, 256, 0, stream>>>(Wk, Wkp);
    pack_w_kernel<<<32, 256, 0, stream>>>(Wv, Wvp);
    pack_w_kernel<<<32, 256, 0, stream>>>(Wo, Wop);

    qkv_kernel<<<rowTiles, 256, 0, stream>>>(xbf, Wqp, Wkp, Wvp,
                                             bq, bk, bv, Q, K, V, n);
    edge_kernel<<<edgeBlocks, 256, 0, stream>>>(src, dst, Q, K, V, wV, z, E);
    attn_ln1_kernel<<<nodeBlocks, 256, 0, stream>>>(x, wV, z, gamma1, beta1, h1, n);
    ln2_kernel<<<nodeBlocks, 256, 0, stream>>>(h1, gamma2, beta2, gbf, n);
    ffn_kernel<<<rowTiles, 256, 0, stream>>>(gbf, h1, Wop, bo, (float*)d_out, n);
}
